// GraphAutoEncoder_55379308314958
// MI455X (gfx1250) — compile-verified, matrix operand order checked by reading, not run on hardware
//
#include <hip/hip_runtime.h>
#include <hip/hip_bf16.h>

// ---------------------------------------------------------------------------
// GCN autoencoder on gfx1250 (MI455X).
//  - Dense per-node GEMMs: v_wmma_f32_16x16x32_bf16, W pre-swizzled into LDS
//    B-fragment layout (ds_load_b128 per operand), A via global_load_b128,
//    one wave owns a full 16xM output strip (M/16 accumulators).
//  - Edge scatter-add via global float atomics (resolve in 192MB L2).
// ---------------------------------------------------------------------------

typedef __attribute__((ext_vector_type(16))) __bf16 v16bf;
typedef __attribute__((ext_vector_type(8)))  float  v8f;

#define NN 50000   // N_NODES (fixed by reference; 50000 % 16 == 0)

// ---------------- degree / normalization precompute ----------------

__global__ void k_deg(const long long* __restrict__ dst,
                      const float* __restrict__ w,
                      float* __restrict__ deg_w, float* __restrict__ deg_1, int E) {
    int e = blockIdx.x * blockDim.x + threadIdx.x;
    if (e >= E) return;
    int t = (int)dst[e];
    atomicAdd(deg_w + t, w[e]);
    atomicAdd(deg_1 + t, 1.0f);
}

__global__ void k_dinv(float* __restrict__ dw, float* __restrict__ d1, int N) {
    int i = blockIdx.x * blockDim.x + threadIdx.x;
    if (i >= N) return;
    dw[i] = rsqrtf(dw[i] + 1.0f);   // +1 self-loop
    d1[i] = rsqrtf(d1[i] + 1.0f);
}

__global__ void k_coef(const long long* __restrict__ src,
                       const long long* __restrict__ dst,
                       const float* __restrict__ w,
                       const float* __restrict__ dinv_w,
                       const float* __restrict__ dinv_1,
                       float* __restrict__ coef_w, float* __restrict__ coef_1, int E) {
    int e = blockIdx.x * blockDim.x + threadIdx.x;
    if (e >= E) return;
    int s = (int)src[e], t = (int)dst[e];
    coef_w[e] = dinv_w[s] * w[e] * dinv_w[t];
    coef_1[e] = dinv_1[s] * dinv_1[t];
}

// ---------------- dense GEMM: H[N,M] = X[N,K] @ W[K,M] ----------------------
// Compile-time K,M. Block = 256 threads (8 waves). W is staged once per block
// into LDS, pre-swizzled into the exact WMMA B-fragment lane layout, so each
// lane reads its 16 bf16 B-operands as two ds_load_b128. Each wave computes a
// full 16xM strip: A fragment loaded once per k-chunk, reused by M/16 WMMAs.

template<int K, int M>
__global__ __launch_bounds__(256) void k_gemm_wmma(const float* __restrict__ X,
                                                   const float* __restrict__ Wm,
                                                   float* __restrict__ H) {
    constexpr int KC = K / 32;   // k-chunks of 32
    constexpr int CT = M / 16;   // 16-wide column tiles
    __shared__ __align__(32) __bf16 bfrag[CT * KC * 32 * 16];

    // Cooperative stage of W into B-fragment order:
    //   flat i = ((ct*KC + c)*32 + lane)*16 + e
    //   lane l holds B column ct*16 + (l&15); element e holds
    //   K index c*32 + (e>=8 ? 16 : 0) + (l>>4)*8 + (e&7)   (ISA 7.12.2)
    for (int i = threadIdx.x; i < K * M; i += 256) {
        const int e  = i & 15;
        const int l  = (i >> 4) & 31;
        const int f  = i >> 9;
        const int c  = f % KC;
        const int ct = f / KC;
        const int k  = c * 32 + ((e & 8) << 1) + ((l >> 4) << 3) + (e & 7);
        const int m  = ct * 16 + (l & 15);
        bfrag[i] = (__bf16)Wm[k * M + m];
    }
    __syncthreads();

    const int lane    = threadIdx.x & 31;
    const int wave    = threadIdx.x >> 5;
    const int rowTile = blockIdx.x * 8 + wave;
    if (rowTile >= NN / 16) return;        // uniform per wave -> EXEC all-1

    const int r    = lane & 15;
    const int half = lane >> 4;
    const float* __restrict__ xrow = X + (size_t)(rowTile * 16 + r) * K;

    v8f acc[CT];
    #pragma unroll
    for (int ct = 0; ct < CT; ++ct) acc[ct] = (v8f){};

    #pragma unroll
    for (int c = 0; c < KC; ++c) {
        const int kb = c * 32 + (half << 3);
        // A fragment: two contiguous 8-float runs -> 4x global_load_b128
        const float4 a0 = *(const float4*)(xrow + kb);
        const float4 a1 = *(const float4*)(xrow + kb + 4);
        const float4 a2 = *(const float4*)(xrow + kb + 16);
        const float4 a3 = *(const float4*)(xrow + kb + 20);
        v16bf a;
        a[0]  = (__bf16)a0.x; a[1]  = (__bf16)a0.y; a[2]  = (__bf16)a0.z; a[3]  = (__bf16)a0.w;
        a[4]  = (__bf16)a1.x; a[5]  = (__bf16)a1.y; a[6]  = (__bf16)a1.z; a[7]  = (__bf16)a1.w;
        a[8]  = (__bf16)a2.x; a[9]  = (__bf16)a2.y; a[10] = (__bf16)a2.z; a[11] = (__bf16)a2.w;
        a[12] = (__bf16)a3.x; a[13] = (__bf16)a3.y; a[14] = (__bf16)a3.z; a[15] = (__bf16)a3.w;

        #pragma unroll
        for (int ct = 0; ct < CT; ++ct) {
            const v16bf b = *(const v16bf*)&bfrag[((ct * KC + c) * 32 + lane) * 16];
            acc[ct] = __builtin_amdgcn_wmma_f32_16x16x32_bf16(
                          /*neg_a=*/false, a, /*neg_b=*/false, b,
                          /*c_mod=*/(short)0, acc[ct],
                          /*reuse_a=*/false, /*reuse_b=*/false);
        }
    }

    // D layout: lane holds col r; VGPR v holds row v + half*8
    #pragma unroll
    for (int ct = 0; ct < CT; ++ct) {
        #pragma unroll
        for (int v = 0; v < 8; ++v) {
            const int row = rowTile * 16 + v + (half << 3);
            H[(size_t)row * M + ct * 16 + r] = acc[ct][v];
        }
    }
}

// ---------------- edge scatter: agg[dst] += coef * H[src] -------------------

__global__ void k_scatter(const float* __restrict__ H,
                          const float* __restrict__ coef,
                          const long long* __restrict__ src,
                          const long long* __restrict__ dst,
                          float* __restrict__ agg, int E, int D) {
    int e = blockIdx.x * blockDim.x + threadIdx.x;
    if (e >= E) return;
    const int d0 = blockIdx.y << 2;       // 4 features per thread
    const float c = coef[e];
    const int s = (int)src[e], t = (int)dst[e];
    const float4 hv = *(const float4*)(H + (long long)s * D + d0);
    float* a = agg + (long long)t * D + d0;
    atomicAdd(a + 0, hv.x * c);
    atomicAdd(a + 1, hv.y * c);
    atomicAdd(a + 2, hv.z * c);
    atomicAdd(a + 3, hv.w * c);
}

// ---------------- combine: out = agg + h*dinv^2 + b (+ relu) ----------------

__global__ void k_combine(const float* __restrict__ agg,
                          const float* __restrict__ h,
                          const float* __restrict__ dinv,
                          const float* __restrict__ bias,
                          float* __restrict__ out,
                          int D, long long total, int relu) {
    long long i = (long long)blockIdx.x * blockDim.x + threadIdx.x;
    if (i >= total) return;
    const int n = (int)(i / D);
    const int d = (int)(i - (long long)n * D);
    const float di = dinv[n];
    float v = agg[i] + h[i] * di * di + bias[d];
    out[i] = relu ? fmaxf(v, 0.0f) : v;
}

// ---------------- host-side layer driver ------------------------------------

template<int K, int M>
static void run_gcn_layer(const float* in, const float* Wm, const float* bias,
                          const float* coef, const float* dinv,
                          const long long* src, const long long* dst, int E,
                          float* h, float* agg, float* out, int relu,
                          hipStream_t stream) {
    const int rowTiles = NN / 16;                        // 3125
    k_gemm_wmma<K, M><<<(rowTiles + 7) / 8, 256, 0, stream>>>(in, Wm, h);
    hipMemsetAsync(agg, 0, (size_t)NN * M * sizeof(float), stream);
    dim3 sg((E + 255) / 256, M / 4);
    k_scatter<<<sg, 256, 0, stream>>>(h, coef, src, dst, agg, E, M);
    const long long total = (long long)NN * M;
    k_combine<<<(int)((total + 255) / 256), 256, 0, stream>>>(
        agg, h, dinv, bias, out, M, total, relu);
}

extern "C" void kernel_launch(void* const* d_in, const int* in_sizes, int n_in,
                              void* d_out, int out_size, void* d_ws, size_t ws_size,
                              hipStream_t stream) {
    // inputs (setup_inputs order)
    const float*     x   = (const float*)d_in[0];        // [N,256]
    const long long* ei  = (const long long*)d_in[1];    // [2,E] int64
    const float*     ew  = (const float*)d_in[2];        // [E]
    const float*     W1  = (const float*)d_in[3];  const float* b1 = (const float*)d_in[4];
    const float*     W2  = (const float*)d_in[5];  const float* b2 = (const float*)d_in[6];
    const float*     W3  = (const float*)d_in[7];  const float* b3 = (const float*)d_in[8];
    const float*     W4  = (const float*)d_in[9];  const float* b4 = (const float*)d_in[10];
    const int E = in_sizes[2];
    const long long* src = ei;
    const long long* dst = ei + E;

    // workspace layout (floats)
    float* ws     = (float*)d_ws;
    float* dinv_w = ws;                       // N
    float* dinv_1 = dinv_w + NN;              // N
    float* coef_w = dinv_1 + NN;              // E
    float* coef_1 = coef_w + E;               // E
    float* bufH   = coef_1 + E;               // N*256  (GEMM result h)
    float* bufA   = bufH + (size_t)NN * 256;  // N*256  (scatter accumulator)
    float* bufC   = bufA + (size_t)NN * 256;  // N*64   (inter-layer activations)

    float* recon = (float*)d_out;             // [N,256]
    float* z     = recon + (size_t)NN * 256;  // [N,32]

    // ---- preprocessing: degrees -> dinv -> per-edge coefficients ----
    hipMemsetAsync(dinv_w, 0, 2 * (size_t)NN * sizeof(float), stream);
    k_deg <<<(E + 255) / 256, 256, 0, stream>>>(dst, ew, dinv_w, dinv_1, E);
    k_dinv<<<(NN + 255) / 256, 256, 0, stream>>>(dinv_w, dinv_1, NN);
    k_coef<<<(E + 255) / 256, 256, 0, stream>>>(src, dst, ew, dinv_w, dinv_1,
                                                coef_w, coef_1, E);

    // ---- layer 1: x[N,256] -> relu -> bufC[N,64] ----
    run_gcn_layer<256, 64>(x, W1, b1, coef_w, dinv_w, src, dst, E,
                           bufH, bufA, bufC, /*relu=*/1, stream);
    // ---- layer 2: bufC[N,64] -> relu -> z[N,32] (output #2) ----
    run_gcn_layer<64, 32>(bufC, W2, b2, coef_w, dinv_w, src, dst, E,
                          bufH, bufA, z, /*relu=*/1, stream);
    // ---- layer 3: z[N,32] -> relu -> bufC[N,64] ----
    run_gcn_layer<32, 64>(z, W3, b3, coef_w, dinv_w, src, dst, E,
                          bufH, bufA, bufC, /*relu=*/1, stream);
    // ---- layer 4 (ones weights): bufC[N,64] -> recon[N,256] (output #1) ----
    run_gcn_layer<64, 256>(bufC, W4, b4, coef_1, dinv_1, src, dst, E,
                           bufH, bufA, recon, /*relu=*/0, stream);
}
